// ObjectMeanDirectAttention_12953621365068
// MI455X (gfx1250) — matrix operand order repaired and checked by reading, compiled, not run on hardware
//
#include <hip/hip_runtime.h>

typedef __attribute__((ext_vector_type(16))) _Float16 v16h;
typedef __attribute__((ext_vector_type(8)))  _Float16 v8h;
typedef __attribute__((ext_vector_type(4)))  _Float16 v4h;
typedef __attribute__((ext_vector_type(8)))  float    v8f;

// ---------------------------------------------------------------------------
// CDNA5 v_wmma_f32_16x16x32_f16 (wave32) layouts:
// A 16x32 f16: lane l -> row r=l&15, kb=(l<16)?0:8; elem e<8 -> K=kb+e,
//              elem e>=8 -> K=kb+16+(e-8).  (two contiguous 8-half runs)
// B 32x16 f16: lane l -> col n=l&15, kb=(l<16)?0:8; elem e -> K as for A.
// C/D 16x16 f32: lane l -> col n=l&15; VGPR j -> row m=j+((l<16)?0:8).
// ---------------------------------------------------------------------------
__device__ __forceinline__ v8f wmma_f16(v16h a, v16h b, v8f c) {
  return __builtin_amdgcn_wmma_f32_16x16x32_f16(false, a, false, b, (short)0, c,
                                                false, false);
}

// Vectorized A-fragment load from f16 LDS tile; stride must be a multiple of
// 8 halves and the base 16B-aligned -> lowers to 2 x ds_load_b128.
__device__ __forceinline__ v16h load_A_frag_v(const _Float16* src, int stride,
                                              int k0, int lane) {
  const int r  = lane & 15;
  const int kb = (lane < 16) ? 0 : 8;
  const _Float16* p = src + r * stride + k0 + kb;
  const v8h lo = *(const v8h*)(p);
  const v8h hi = *(const v8h*)(p + 16);
  v16h a;
#pragma unroll
  for (int e = 0; e < 8; ++e) {
    a[e] = lo[e];
    a[e + 8] = hi[e];
  }
  return a;
}

// ---------------------------------------------------------------------------
// Weight staging pre-swizzled into B-fragment order:
//   dst[(frag*32 + lane)*16 + e] = (f16) W[k][n],
//   frag = ct*ksteps + ks, n = ct*16 + (lane&15),
//   k = ks*32 + ((lane<16)?0:8) + (e&7) + ((e>>3)<<4).
// B fragment load is then 2 x aligned ds_load_b128 per lane.
// W is row-major [K][64] f32 in global memory. Cooperative across the block.
// ---------------------------------------------------------------------------
__device__ inline void stage_B_swz(_Float16* dst, const float* __restrict__ w,
                                   int K, int tid, int nthreads) {
  const int ksteps = K >> 5;            // K / 32
  const int total  = 4 * ksteps * 32 * 16;
  for (int i = tid; i < total; i += nthreads) {
    const int e    = i & 15;
    const int lane = (i >> 4) & 31;
    const int f    = i >> 9;
    const int ct   = f / ksteps;
    const int ks   = f - ct * ksteps;
    const int n    = ct * 16 + (lane & 15);
    const int kb   = (lane < 16) ? 0 : 8;
    const int k    = ks * 32 + kb + (e & 7) + ((e >> 3) << 4);
    dst[i] = (_Float16)w[k * 64 + n];
  }
}

__device__ __forceinline__ v16h load_B_frag_swz(const _Float16* base, int f,
                                                int lane) {
  const _Float16* p = base + (((f << 5) + lane) << 4);
  const v8h lo = *(const v8h*)(p);
  const v8h hi = *(const v8h*)(p + 8);
  v16h b;
#pragma unroll
  for (int e = 0; e < 8; ++e) {
    b[e] = lo[e];
    b[e + 8] = hi[e];
  }
  return b;
}

// ---------------------------------------------------------------------------
// Kernel 1: zero the workspace accumulators
// ---------------------------------------------------------------------------
__global__ void zero_ws(float* __restrict__ p, size_t n) {
  size_t i = (size_t)blockIdx.x * blockDim.x + threadIdx.x;
  size_t stride = (size_t)gridDim.x * blockDim.x;
  for (; i < n; i += stride) p[i] = 0.0f;
}

// ---------------------------------------------------------------------------
// Kernel 2: per-node attention MLP (64->64->64->1, ReLU) + weighted scatter.
// 128 threads = 4 wave32s; each wave owns 16-node tiles. Weight fragments
// live in registers (staged once via swizzled LDS).
// ---------------------------------------------------------------------------
#define WAVES_B 4
#define XSTRF 68  // f32 tile row stride (floats): bank-conflict-free columns
#define XSTRH 72  // f16 tile row stride (halves): mult of 8 -> b128 frag loads

__global__ void __launch_bounds__(128)
node_attn_scatter(const float* __restrict__ x, const int* __restrict__ batch,
                  const float* __restrict__ W1, const float* __restrict__ b1,
                  const float* __restrict__ W2, const float* __restrict__ b2,
                  const float* __restrict__ W3, const float* __restrict__ b3,
                  float* __restrict__ num, float* __restrict__ cnt,
                  int tilesTotal) {
  __shared__ __align__(16) _Float16 sW1s[8 * 32 * 16];          // swizzled W1
  __shared__ __align__(16) _Float16 sW2s[8 * 32 * 16];          // swizzled W2
  __shared__ float sxb[WAVES_B][16 * XSTRF];                    // x f32 (scatter)
  __shared__ __align__(16) _Float16 sxh[WAVES_B][16 * XSTRH];   // x f16 (A frags)
  __shared__ __align__(16) _Float16 shh[WAVES_B][16 * XSTRH];   // h1 f16

  const int tid  = threadIdx.x;
  const int wave = tid >> 5;
  const int lane = tid & 31;
  const int ncol = lane & 15;
  const int mh   = (lane < 16) ? 0 : 8;
  float* xb    = sxb[wave];
  _Float16* xh = sxh[wave];
  _Float16* hh = shh[wave];

  // ---- cooperative swizzled weight staging, then lift frags to registers ----
  stage_B_swz(sW1s, W1, 64, tid, blockDim.x);
  stage_B_swz(sW2s, W2, 64, tid, blockDim.x);
  __syncthreads();

  v16h w1f[4][2], w2f[4][2];
  float bias1[4], bias2[4], w3v[4];
#pragma unroll
  for (int ct = 0; ct < 4; ++ct) {
#pragma unroll
    for (int ks = 0; ks < 2; ++ks) {
      w1f[ct][ks] = load_B_frag_swz(sW1s, ct * 2 + ks, lane);
      w2f[ct][ks] = load_B_frag_swz(sW2s, ct * 2 + ks, lane);
    }
    bias1[ct] = b1[ct * 16 + ncol];
    bias2[ct] = b2[ct * 16 + ncol];
    w3v[ct]   = W3[ct * 16 + ncol];
  }
  const float b3v = b3[0];

  for (int tile = blockIdx.x * WAVES_B + wave; tile < tilesTotal;
       tile += gridDim.x * WAVES_B) {
    const int base = tile * 16;

    // --- stage x tile: f32 copy (scatter) + f16 copy (fragments) ---
#pragma unroll
    for (int v = 0; v < 8; ++v) {
      const int linear = v * 128 + lane * 4;
      const int m = linear >> 6;
      const int f = linear & 63;
      const float4 val = *(const float4*)(x + (size_t)(base + m) * 64 + f);
      *(float4*)(xb + m * XSTRF + f) = val;
      const v4h hv = {(_Float16)val.x, (_Float16)val.y, (_Float16)val.z,
                      (_Float16)val.w};
      *(v4h*)(xh + m * XSTRH + f) = hv;
    }
    const int gv = batch[base + ncol];  // lane&15 -> node's graph id

    // --- layer 1: h1 = relu(x @ W1 + b1) -> LDS f16 ---
    const v16h a0 = load_A_frag_v(xh, XSTRH, 0, lane);
    const v16h a1 = load_A_frag_v(xh, XSTRH, 32, lane);
#pragma unroll
    for (int ct = 0; ct < 4; ++ct) {
      v8f c = {};
      c = wmma_f16(a0, w1f[ct][0], c);
      c = wmma_f16(a1, w1f[ct][1], c);
#pragma unroll
      for (int j = 0; j < 8; ++j) {
        hh[(j + mh) * XSTRH + ct * 16 + ncol] =
            (_Float16)fmaxf(c[j] + bias1[ct], 0.0f);
      }
    }

    // --- layer 2 fused with layer 3 partial dot (all in registers) ---
    const v16h h0 = load_A_frag_v(hh, XSTRH, 0, lane);
    const v16h h1 = load_A_frag_v(hh, XSTRH, 32, lane);
    float part[8];
#pragma unroll
    for (int j = 0; j < 8; ++j) part[j] = 0.0f;
#pragma unroll
    for (int ct = 0; ct < 4; ++ct) {
      v8f c = {};
      c = wmma_f16(h0, w2f[ct][0], c);
      c = wmma_f16(h1, w2f[ct][1], c);
#pragma unroll
      for (int j = 0; j < 8; ++j) {
        const float h2 = fmaxf(c[j] + bias2[ct], 0.0f);
        part[j] += h2 * w3v[ct];  // this lane's column contribution
      }
    }
    // butterfly-reduce across the 16 columns held by each lane half,
    // then exchange halves so every lane holds all 16 per-node scalars.
    float alo[8], ahi[8];
#pragma unroll
    for (int j = 0; j < 8; ++j) {
      float p = part[j];
      p += __shfl_xor(p, 1);
      p += __shfl_xor(p, 2);
      p += __shfl_xor(p, 4);
      p += __shfl_xor(p, 8);
      p += b3v;  // a[j+mh] complete on this lane
      const float o = __shfl_xor(p, 16);
      alo[j] = (mh == 0) ? p : o;  // a[j]
      ahi[j] = (mh == 0) ? o : p;  // a[j+8]
    }

    // --- run-compacted attention-weighted scatter (batch is sorted) ---
    float acc0 = 0.0f, acc1 = 0.0f;
    int curg = __shfl(gv, 0);
    int run  = 0;
#pragma unroll
    for (int m = 0; m < 16; ++m) {
      const int g = __shfl(gv, m);
      if (g != curg) {
        atomicAdd(&num[(size_t)curg * 64 + lane], acc0);
        atomicAdd(&num[(size_t)curg * 64 + 32 + lane], acc1);
        if (lane == 0) atomicAdd(&cnt[curg], (float)run);
        acc0 = acc1 = 0.0f;
        run = 0;
        curg = g;
      }
      const float am = (m < 8) ? alo[m] : ahi[m - 8];
      acc0 += xb[m * XSTRF + lane] * am;
      acc1 += xb[m * XSTRF + 32 + lane] * am;
      ++run;
    }
    atomicAdd(&num[(size_t)curg * 64 + lane], acc0);
    atomicAdd(&num[(size_t)curg * 64 + 32 + lane], acc1);
    if (lane == 0) atomicAdd(&cnt[curg], (float)run);
  }
}

// ---------------------------------------------------------------------------
// Kernel 3: u = [num1/cnt1, num2/cnt2]; out = MLP(u): 128->64->64->64
// Weights staged pre-swizzled -> every B fragment is 2 x ds_load_b128.
// ---------------------------------------------------------------------------
#define WAVES_C 4
#define USTR 136  // mult of 8 halves -> vector A-frag loads
#define HSTR 72

__global__ void __launch_bounds__(128)
finalize_mlp(const float* __restrict__ num1, const float* __restrict__ cnt1,
             const float* __restrict__ num2, const float* __restrict__ cnt2,
             const float* __restrict__ W1, const float* __restrict__ b1,
             const float* __restrict__ W2, const float* __restrict__ b2,
             const float* __restrict__ W3, const float* __restrict__ b3,
             float* __restrict__ out, int Bg) {
  __shared__ __align__(16) _Float16 sW1s[16 * 32 * 16];  // 16 frags (K=128)
  __shared__ __align__(16) _Float16 sW2s[8 * 32 * 16];
  __shared__ __align__(16) _Float16 sW3s[8 * 32 * 16];
  __shared__ float sb1[64], sb2[64], sb3[64];
  __shared__ __align__(16) _Float16 sub[WAVES_C][16 * USTR];
  __shared__ __align__(16) _Float16 shb[WAVES_C][16 * HSTR];

  const int tid = threadIdx.x;
  stage_B_swz(sW1s, W1, 128, tid, blockDim.x);
  stage_B_swz(sW2s, W2, 64, tid, blockDim.x);
  stage_B_swz(sW3s, W3, 64, tid, blockDim.x);
  for (int i = tid; i < 64; i += blockDim.x) {
    sb1[i] = b1[i];
    sb2[i] = b2[i];
    sb3[i] = b3[i];
  }
  __syncthreads();

  const int wave = tid >> 5;
  const int lane = tid & 31;
  _Float16* ub = sub[wave];
  _Float16* hb = shb[wave];
  const int ncol = lane & 15;
  const int mh   = (lane < 16) ? 0 : 8;
  const int tilesTotal = Bg / 16;

  for (int tile = blockIdx.x * WAVES_C + wave; tile < tilesTotal;
       tile += gridDim.x * WAVES_C) {
    const int base = tile * 16;

    // --- build u tile: 16 graphs x 128 cols, f16 ---
#pragma unroll 4
    for (int v = 0; v < 64; ++v) {
      const int linear = v * 32 + lane;
      const int m = linear >> 7;
      const int f = linear & 127;
      const int g = base + m;
      float val;
      if (f < 64)
        val = num1[(size_t)g * 64 + f] / fmaxf(cnt1[g], 1.0f);
      else
        val = num2[(size_t)g * 64 + (f - 64)] / fmaxf(cnt2[g], 1.0f);
      ub[m * USTR + f] = (_Float16)val;
    }

    // --- layer 1: K = 128, 4 k-steps ---
    const v16h u0 = load_A_frag_v(ub, USTR, 0, lane);
    const v16h u1 = load_A_frag_v(ub, USTR, 32, lane);
    const v16h u2 = load_A_frag_v(ub, USTR, 64, lane);
    const v16h u3 = load_A_frag_v(ub, USTR, 96, lane);
#pragma unroll
    for (int ct = 0; ct < 4; ++ct) {
      v8f c = {};
      c = wmma_f16(u0, load_B_frag_swz(sW1s, ct * 4 + 0, lane), c);
      c = wmma_f16(u1, load_B_frag_swz(sW1s, ct * 4 + 1, lane), c);
      c = wmma_f16(u2, load_B_frag_swz(sW1s, ct * 4 + 2, lane), c);
      c = wmma_f16(u3, load_B_frag_swz(sW1s, ct * 4 + 3, lane), c);
      const float bias = sb1[ct * 16 + ncol];
#pragma unroll
      for (int j = 0; j < 8; ++j) {
        hb[(j + mh) * HSTR + ct * 16 + ncol] =
            (_Float16)fmaxf(c[j] + bias, 0.0f);
      }
    }

    // --- layer 2 ---
    {
      const v16h h0 = load_A_frag_v(hb, HSTR, 0, lane);
      const v16h h1 = load_A_frag_v(hb, HSTR, 32, lane);
#pragma unroll
      for (int ct = 0; ct < 4; ++ct) {
        v8f c = {};
        c = wmma_f16(h0, load_B_frag_swz(sW2s, ct * 2 + 0, lane), c);
        c = wmma_f16(h1, load_B_frag_swz(sW2s, ct * 2 + 1, lane), c);
        const float bias = sb2[ct * 16 + ncol];
#pragma unroll
        for (int j = 0; j < 8; ++j) {
          hb[(j + mh) * HSTR + ct * 16 + ncol] =
              (_Float16)fmaxf(c[j] + bias, 0.0f);
        }
      }
    }

    // --- layer 3 (no relu), write f32 output ---
    {
      const v16h h0 = load_A_frag_v(hb, HSTR, 0, lane);
      const v16h h1 = load_A_frag_v(hb, HSTR, 32, lane);
#pragma unroll
      for (int ct = 0; ct < 4; ++ct) {
        v8f c = {};
        c = wmma_f16(h0, load_B_frag_swz(sW3s, ct * 2 + 0, lane), c);
        c = wmma_f16(h1, load_B_frag_swz(sW3s, ct * 2 + 1, lane), c);
        const float bias = sb3[ct * 16 + ncol];
#pragma unroll
        for (int j = 0; j < 8; ++j) {
          out[(size_t)(base + j + mh) * 64 + ct * 16 + ncol] = c[j] + bias;
        }
      }
    }
  }
}

// ---------------------------------------------------------------------------
extern "C" void kernel_launch(void* const* d_in, const int* in_sizes, int n_in,
                              void* d_out, int out_size, void* d_ws,
                              size_t ws_size, hipStream_t stream) {
  const float* x1     = (const float*)d_in[0];
  const int*   batch1 = (const int*)d_in[1];
  const float* x2     = (const float*)d_in[2];
  const int*   batch2 = (const int*)d_in[3];
  const float* aW1 = (const float*)d_in[4];
  const float* ab1 = (const float*)d_in[5];
  const float* aW2 = (const float*)d_in[6];
  const float* ab2 = (const float*)d_in[7];
  const float* aW3 = (const float*)d_in[8];
  const float* ab3 = (const float*)d_in[9];
  const float* fW1 = (const float*)d_in[10];
  const float* fb1 = (const float*)d_in[11];
  const float* fW2 = (const float*)d_in[12];
  const float* fb2 = (const float*)d_in[13];
  const float* fW3 = (const float*)d_in[14];
  const float* fb3 = (const float*)d_in[15];
  float* out = (float*)d_out;

  const int N  = in_sizes[0] / 64;  // nodes per input (1,000,000)
  const int Bg = out_size / 64;     // graphs (4096)

  float* ws   = (float*)d_ws;
  float* num1 = ws;
  float* num2 = ws + (size_t)Bg * 64;
  float* cnt1 = ws + (size_t)Bg * 128;
  float* cnt2 = cnt1 + Bg;
  const size_t zn = (size_t)Bg * 128 + 2 * (size_t)Bg;

  zero_ws<<<512, 256, 0, stream>>>(ws, zn);

  const int tiles = N / 16;  // N divisible by 16 for this workload
  node_attn_scatter<<<1024, 128, 0, stream>>>(x1, batch1, aW1, ab1, aW2, ab2,
                                              aW3, ab3, num1, cnt1, tiles);
  node_attn_scatter<<<1024, 128, 0, stream>>>(x2, batch2, aW1, ab1, aW2, ab2,
                                              aW3, ab3, num2, cnt2, tiles);

  const int ctiles  = Bg / 16;
  const int cblocks = (ctiles + WAVES_C - 1) / WAVES_C;
  finalize_mlp<<<cblocks, 128, 0, stream>>>(num1, cnt1, num2, cnt2, fW1, fb1,
                                            fW2, fb2, fW3, fb3, out, Bg);
}
